// Qwen3MoeAttention_721554505835
// MI455X (gfx1250) — matrix operand order repaired and checked by reading, not compiled, not run on hardware
//
#include <hip/hip_runtime.h>
#include <hip/hip_bf16.h>

// ---------------------------------------------------------------------------
// Qwen3 attention block for gfx1250 (MI455X), bf16 WMMA pipeline.
//   B=1, S=2048, H=2048, NH=32, NKV=4, HD=128, G=8
// ---------------------------------------------------------------------------

typedef __bf16 bf16;
typedef __attribute__((ext_vector_type(16))) __bf16 v16bf;
typedef __attribute__((ext_vector_type(8)))  __bf16 v8bf;
typedef __attribute__((ext_vector_type(8)))  float  v8f;
typedef __attribute__((ext_vector_type(4)))  unsigned v4u;
typedef __attribute__((ext_vector_type(8)))  unsigned v8u;

#define S_LEN 2048
#define H_DIM 2048
#define NH    32
#define NKV   4
#define HD    128
#define EPSF  1e-6f
// 1/sqrt(128)
#define SCALE 0.08838834764831845f
#define NEG_BIG -3.402823466e38f

static __device__ __forceinline__ v16bf frag_cat(v8bf a, v8bf b) {
    return __builtin_shufflevector(a, b, 0,1,2,3,4,5,6,7,8,9,10,11,12,13,14,15);
}

static __device__ __forceinline__ v8f wmma_bf16(v16bf a, v16bf b, v8f c) {
    // D = A(16x32 bf16) * B(32x16 bf16) + C(16x16 f32)
    return __builtin_amdgcn_wmma_f32_16x16x32_bf16(
        /*neg_a=*/false, a, /*neg_b=*/false, b,
        /*c_mod=*/(short)0, c, /*reuse_a=*/false, /*reuse_b=*/false);
}

// Async global->LDS 16B copy (ASYNCcnt path). Low 32 bits of a generic LDS
// pointer are the wave-relative LDS byte address (flat->LDS drops high bits).
static __device__ __forceinline__ void async_copy_b128(const bf16* gsrc, bf16* ldst) {
    unsigned loff = (unsigned)(unsigned long long)ldst;
    asm volatile("global_load_async_to_lds_b128 %0, %1, off"
                 :: "v"(loff), "v"(gsrc) : "memory");
}
static __device__ __forceinline__ void wait_async() {
    asm volatile("s_wait_asynccnt 0x0" ::: "memory");
}
static __device__ __forceinline__ void wait_ds() {
    asm volatile("s_wait_dscnt 0x0" ::: "memory");
}

// Tensor Data Mover: 2D tile load global->LDS (TENSORcnt path).
//   rows x rowbytes tile, contiguous rows in memory (row stride = rowbytes),
//   LDS rows padded by padbytes per row. data_size = 8B units.
//   rowbytes and padbytes must be multiples of 8 / DWORD-expressible.
static __device__ __forceinline__ void tdm_load_2d(const void* gsrc, void* ldst,
                                                   unsigned rows, unsigned row8 /*8B units*/,
                                                   unsigned pad_interval_code,
                                                   unsigned pad_amount_code) {
    const unsigned long long ga = (unsigned long long)gsrc;
    const unsigned la = (unsigned)(unsigned long long)ldst;
    v4u g0;
    g0[0] = 1u;                                   // count=1, user descriptor
    g0[1] = la;                                   // lds_addr
    g0[2] = (unsigned)ga;                         // global_addr[31:0]
    g0[3] = (unsigned)((ga >> 32) & 0x01FFFFFFu)  // global_addr[56:32]
          | (2u << 30);                           // type=2 (image)
    v8u g1;
    g1[0] = (3u << 16)                            // data_size = 8B
          | (1u << 20)                            // pad_enable
          | (pad_interval_code << 22)             // DWORDs before pad
          | (pad_amount_code << 25);              // pad DWORDs
    g1[1] = (row8 & 0xFFFFu) << 16;               // tensor_dim0[15:0]
    g1[2] = (rows & 0xFFFFu) << 16;               // (dim0 hi=0) | tensor_dim1[15:0]
    g1[3] = (row8 & 0xFFFFu) << 16;               // (dim1 hi=0) | tile_dim0
    g1[4] = (rows & 0xFFFFu);                     // tile_dim1 | (tile_dim2=0)
    g1[5] = row8;                                 // tensor_dim0_stride[31:0]
    g1[6] = 0u;                                   // stride hi | dim1_stride lo
    g1[7] = 0u;
    asm volatile("tensor_load_to_lds %0, %1" :: "s"(g0), "s"(g1) : "memory");
}

// A-fragment (16x32, 16-bit): lane m = lane&15, hi = lane>>4.
//   elements 0..7  : K = hi*8    .. hi*8+7     (contiguous)
//   elements 8..15 : K = 16+hi*8 .. 16+hi*8+7  (contiguous)
// B-fragment (32x16, 16-bit), column-major storage Bt[n][k]:
//   elements 0..7 : K = hi*8 .. hi*8+7, 8..15 : K = 16+hi*8 .. 16+hi*8+7 (same as A)
// C/D (16x16 f32): VGPR r holds row (hi*8 + r), lane col n = lane&15.

// ===========================================================================
// Kernel 1: fused QKV projection + per-head RMSNorm + RoPE (Q,K), bf16 out.
// grid = (S/128, 40): y<32 -> Q head, y<36 -> K head, else V head.
// ===========================================================================
__global__ __launch_bounds__(256)
void qkv_rope_kernel(const float* __restrict__ hs,
                     const float* __restrict__ wq,
                     const float* __restrict__ wk,
                     const float* __restrict__ wv,
                     const float* __restrict__ qnw,
                     const float* __restrict__ knw,
                     const float* __restrict__ cosp,
                     const float* __restrict__ sinp,
                     bf16* __restrict__ q_ws,
                     bf16* __restrict__ k_ws,
                     bf16* __restrict__ v_ws)
{
    const int s0   = blockIdx.x * 128;
    const int head = blockIdx.y;          // 0..39
    const int tid  = threadIdx.x & 255;   // provable bound for staging loops
    const int wave = tid >> 5;
    const int lane = tid & 31;
    const int mlan = lane & 15;
    const int hi   = lane >> 4;

    const float* W; int ldw, n0, oh;
    bool do_norm; const float* nw; bf16* outp;
    if (head < 32)      { W = wq; ldw = NH*HD;  n0 = head*HD;      do_norm = true;  nw = qnw; outp = q_ws; oh = head;      }
    else if (head < 36) { W = wk; ldw = NKV*HD; n0 = (head-32)*HD; do_norm = true;  nw = knw; outp = k_ws; oh = head - 32; }
    else                { W = wv; ldw = NKV*HD; n0 = (head-36)*HD; do_norm = false; nw = qnw; outp = v_ws; oh = head - 36; }

    __shared__ __align__(16) bf16 Alds[128 * 40];   // 128 rows x 32 K, stride 40
    __shared__ __align__(16) bf16 Bt  [128 * 40];   // 128 N   x 32 K, stride 40 (transposed)

    v8f acc[8];
    #pragma unroll
    for (int t = 0; t < 8; ++t) acc[t] = v8f{};

    for (int k0 = 0; k0 < H_DIM; k0 += 32) {
        __syncthreads();
        // stage A: 128 x 32 fp32 -> bf16 (batched float4 loads, then stores)
        {
            float4 f[4];
            #pragma unroll
            for (int j = 0; j < 4; ++j) {
                int i = tid + j * 256, r = i >> 3, c = (i & 7) * 4;
                f[j] = *(const float4*)&hs[(size_t)(s0 + r) * H_DIM + (k0 + c)];
            }
            #pragma unroll
            for (int j = 0; j < 4; ++j) {
                int i = tid + j * 256, r = i >> 3, c = (i & 7) * 4;
                bf16 h4[4] = { (bf16)f[j].x, (bf16)f[j].y, (bf16)f[j].z, (bf16)f[j].w };
                *(uint2*)&Alds[r * 40 + c] = *(const uint2*)h4;
            }
        }
        // stage B transposed: 32 x 128 fp32 -> Bt[n][k]
        {
            float4 f[4];
            #pragma unroll
            for (int j = 0; j < 4; ++j) {
                int i = tid + j * 256, kk = i >> 5, n4 = (i & 31) * 4;
                f[j] = *(const float4*)&W[(size_t)(k0 + kk) * ldw + (n0 + n4)];
            }
            #pragma unroll
            for (int j = 0; j < 4; ++j) {
                int i = tid + j * 256, kk = i >> 5, n4 = (i & 31) * 4;
                Bt[(n4 + 0) * 40 + kk] = (bf16)f[j].x;
                Bt[(n4 + 1) * 40 + kk] = (bf16)f[j].y;
                Bt[(n4 + 2) * 40 + kk] = (bf16)f[j].z;
                Bt[(n4 + 3) * 40 + kk] = (bf16)f[j].w;
            }
        }
        // prefetch next K-step weight tile
        if (k0 + 32 < H_DIM)
            __builtin_prefetch(&W[(size_t)(k0 + 32 + (tid >> 3)) * ldw + n0 + (tid & 7) * 16], 0, 1);
        __syncthreads();

        const bf16* arow = &Alds[(wave * 16 + mlan) * 40];
        v16bf afrag = frag_cat(*(const v8bf*)(arow + hi * 8),
                               *(const v8bf*)(arow + 16 + hi * 8));
        v16bf bfr[8];
        #pragma unroll
        for (int t = 0; t < 8; ++t) {
            const bf16* brow = &Bt[(t * 16 + mlan) * 40];
            bfr[t] = frag_cat(*(const v8bf*)(brow + hi * 8), *(const v8bf*)(brow + 16 + hi * 8));
        }
        #pragma unroll
        for (int t = 0; t < 8; ++t)
            acc[t] = wmma_bf16(afrag, bfr[t], acc[t]);
    }

    // epilogue: RMSNorm + RoPE for Q/K, plain cast for V
    #pragma unroll
    for (int r = 0; r < 8; ++r) {
        const int srow = s0 + wave * 16 + hi * 8 + r;
        float v[8];
        #pragma unroll
        for (int t = 0; t < 8; ++t) v[t] = acc[t][r];

        if (do_norm) {
            float ss = 0.f;
            #pragma unroll
            for (int t = 0; t < 8; ++t) ss += v[t] * v[t];
            #pragma unroll
            for (int m = 1; m < 16; m <<= 1) ss += __shfl_xor(ss, m, 32);
            const float rms = rsqrtf(ss * (1.0f / (float)HD) + EPSF);
            float nv[8];
            #pragma unroll
            for (int t = 0; t < 8; ++t) {
                int d = t * 16 + mlan;
                nv[t] = v[t] * rms * nw[d];
            }
            #pragma unroll
            for (int t = 0; t < 8; ++t) {
                int d = t * 16 + mlan;
                float rot = (t < 4) ? -nv[t + 4] : nv[t - 4];
                v[t] = nv[t] * cosp[(size_t)srow * HD + d] + rot * sinp[(size_t)srow * HD + d];
            }
        }
        const size_t base = ((size_t)oh * S_LEN + srow) * HD;
        #pragma unroll
        for (int t = 0; t < 8; ++t)
            outp[base + t * 16 + mlan] = (bf16)v[t];
    }
}

// ===========================================================================
// Kernel 2: GQA causal flash attention, bf16 in/out, f32 accumulation.
// grid = (S/128, NH). block = 256 (8 waves); wave owns 16 q-rows x HD.
// K tile staged via TDM (tensor_load_to_lds), V tile transposed manually.
// ===========================================================================
__global__ __launch_bounds__(256)
void flash_attn_kernel(const bf16* __restrict__ q_ws,
                       const bf16* __restrict__ k_ws,
                       const bf16* __restrict__ v_ws,
                       bf16* __restrict__ attn_ws)
{
    const int qb   = blockIdx.x;          // 0..15
    const int h    = blockIdx.y;          // 0..31
    const int kv   = h >> 3;              // GQA group
    const int tid  = threadIdx.x & 255;
    const int wave = tid >> 5;
    const int lane = tid & 31;
    const int mlan = lane & 15;
    const int hi   = lane >> 4;
    const int s0   = qb * 128;

    __shared__ __align__(16) bf16 Klds[32 * 136];     // [key][d], stride 136 (=256B+16B pad)
    __shared__ __align__(16) bf16 Vt  [128 * 40];     // [d][key], stride 40
    __shared__ __align__(16) bf16 Plds[8][16 * 40];   // per-wave P, stride 40

    // preload Q fragments (reused every K step)
    v16bf qf[4];
    {
        const int srow = s0 + wave * 16 + mlan;
        const bf16* qrow = q_ws + ((size_t)h * S_LEN + srow) * HD;
        #pragma unroll
        for (int ks = 0; ks < 4; ++ks)
            qf[ks] = frag_cat(*(const v8bf*)(qrow + ks * 32 + hi * 8),
                              *(const v8bf*)(qrow + ks * 32 + 16 + hi * 8));
    }

    v8f oacc[8];
    #pragma unroll
    for (int t = 0; t < 8; ++t) oacc[t] = v8f{};
    float mrow[8], lrow[8];
    #pragma unroll
    for (int r = 0; r < 8; ++r) { mrow[r] = NEG_BIG; lrow[r] = 0.f; }

    const int nkb = (qb + 1) * 4;         // causal: keys <= q-tile end
    for (int kb = 0; kb < nkb; ++kb) {
        __syncthreads();
        const bf16* ksrc = k_ws + ((size_t)kv * S_LEN + kb * 32) * HD;
        const bf16* vsrc = v_ws + ((size_t)kv * S_LEN + kb * 32) * HD;

        // K tile [32 keys x 256B rows] -> Klds with 16B/row padding, via TDM.
        // pad_interval code 5 = 64 DWORDs (256B), pad_amount code 3 = 4 DWORDs (16B).
        if (wave == 0)
            tdm_load_2d(ksrc, Klds, /*rows=*/32, /*row8=*/32,
                        /*pad_interval=*/5u, /*pad_amount=*/3u);

        // stage V transposed: Vt[d][key] (batched vector loads, scatter stores)
        {
            v8bf vv[2];
            #pragma unroll
            for (int j = 0; j < 2; ++j) {
                int i = tid + j * 256, key = i >> 4, dc = (i & 15) * 8;
                vv[j] = *(const v8bf*)(vsrc + (size_t)key * HD + dc);
            }
            #pragma unroll
            for (int j = 0; j < 2; ++j) {
                int i = tid + j * 256, key = i >> 4, dc = (i & 15) * 8;
                #pragma unroll
                for (int e = 0; e < 8; ++e)
                    Vt[(dc + e) * 40 + key] = vv[j][e];
            }
        }
        __builtin_amdgcn_s_wait_tensorcnt(0);
        __syncthreads();

        // scores: S = Q * K^T  (2 tiles of 16x16, K-dim = HD in 4 steps)
        v8f sa[2];
        #pragma unroll
        for (int nt = 0; nt < 2; ++nt) {
            v16bf kf[4];
            #pragma unroll
            for (int ks = 0; ks < 4; ++ks) {
                const bf16* krow = &Klds[(nt * 16 + mlan) * 136 + ks * 32];
                kf[ks] = frag_cat(*(const v8bf*)(krow + hi * 8), *(const v8bf*)(krow + 16 + hi * 8));
            }
            sa[nt] = v8f{};
            #pragma unroll
            for (int ks = 0; ks < 4; ++ks)
                sa[nt] = wmma_bf16(qf[ks], kf[ks], sa[nt]);
        }

        // scale + causal mask
        float sc[2][8];
        #pragma unroll
        for (int nt = 0; nt < 2; ++nt) {
            const int key = kb * 32 + nt * 16 + mlan;
            #pragma unroll
            for (int r = 0; r < 8; ++r) {
                const int row = s0 + wave * 16 + hi * 8 + r;
                float x = sa[nt][r] * SCALE;
                sc[nt][r] = (key <= row) ? x : NEG_BIG;
            }
        }

        // online softmax per row; stash P (bf16) into per-wave LDS
        #pragma unroll
        for (int r = 0; r < 8; ++r) {
            float mx = fmaxf(sc[0][r], sc[1][r]);
            #pragma unroll
            for (int m = 1; m < 16; m <<= 1) mx = fmaxf(mx, __shfl_xor(mx, m, 32));
            const float nm = fmaxf(mrow[r], mx);
            const float p0 = __expf(sc[0][r] - nm);
            const float p1 = __expf(sc[1][r] - nm);
            float ps = p0 + p1;
            #pragma unroll
            for (int m = 1; m < 16; m <<= 1) ps += __shfl_xor(ps, m, 32);
            const float corr = __expf(mrow[r] - nm);
            lrow[r] = lrow[r] * corr + ps;
            mrow[r] = nm;
            #pragma unroll
            for (int t = 0; t < 8; ++t) oacc[t][r] *= corr;
            bf16* prow = &Plds[wave][(hi * 8 + r) * 40];
            prow[mlan]      = (bf16)p0;
            prow[16 + mlan] = (bf16)p1;
        }

        // wave-private LDS write->read ordering
        wait_ds();

        // O += P (16x32) * V (32x128), V-fragments preloaded in groups of 4
        const bf16* pr = &Plds[wave][mlan * 40];
        v16bf pf = frag_cat(*(const v8bf*)(pr + hi * 8),
                            *(const v8bf*)(pr + 16 + hi * 8));
        #pragma unroll
        for (int tg = 0; tg < 2; ++tg) {
            v16bf vf[4];
            #pragma unroll
            for (int t = 0; t < 4; ++t) {
                const bf16* vr = &Vt[((tg * 4 + t) * 16 + mlan) * 40];
                vf[t] = frag_cat(*(const v8bf*)(vr + hi * 8), *(const v8bf*)(vr + 16 + hi * 8));
            }
            #pragma unroll
            for (int t = 0; t < 4; ++t)
                oacc[tg * 4 + t] = wmma_bf16(pf, vf[t], oacc[tg * 4 + t]);
        }
    }

    // finalize: O /= l, store bf16 to attn_ws[s][h*HD + d]
    #pragma unroll
    for (int r = 0; r < 8; ++r) {
        const float inv = 1.0f / lrow[r];
        const int row = s0 + wave * 16 + hi * 8 + r;
        #pragma unroll
        for (int t = 0; t < 8; ++t)
            attn_ws[(size_t)row * (NH * HD) + h * HD + t * 16 + mlan] =
                (bf16)(oacc[t][r] * inv);
    }
}

// ===========================================================================
// Kernel 3: output projection  out = attn (2048x4096 bf16) @ wo (4096x2048)
// grid = (S/128, H/128), block = 256 (8 waves); A tile via async-LDS loads.
// ===========================================================================
__global__ __launch_bounds__(256)
void out_proj_kernel(const bf16* __restrict__ attn_ws,
                     const float* __restrict__ wo,
                     float* __restrict__ out)
{
    const int s0   = blockIdx.x * 128;
    const int n0   = blockIdx.y * 128;
    const int tid  = threadIdx.x & 255;
    const int wave = tid >> 5;
    const int lane = tid & 31;
    const int mlan = lane & 15;
    const int hi   = lane >> 4;

    __shared__ __align__(16) bf16 Alds[128 * 40];
    __shared__ __align__(16) bf16 Bt  [128 * 40];

    v8f acc[8];
    #pragma unroll
    for (int t = 0; t < 8; ++t) acc[t] = v8f{};

    for (int k0 = 0; k0 < NH * HD; k0 += 32) {
        __syncthreads();
        // stage A (already bf16): async global->LDS 16B copies (ASYNCcnt)
        #pragma unroll
        for (int j = 0; j < 2; ++j) {
            int i = tid + j * 256, r = i >> 2, c = (i & 3) * 8;
            async_copy_b128(&attn_ws[(size_t)(s0 + r) * (NH * HD) + (k0 + c)],
                            &Alds[r * 40 + c]);
        }
        // stage B transposed: 32 x 128 fp32 -> Bt[n][k]
        {
            float4 f[4];
            #pragma unroll
            for (int j = 0; j < 4; ++j) {
                int i = tid + j * 256, kk = i >> 5, n4 = (i & 31) * 4;
                f[j] = *(const float4*)&wo[(size_t)(k0 + kk) * H_DIM + (n0 + n4)];
            }
            #pragma unroll
            for (int j = 0; j < 4; ++j) {
                int i = tid + j * 256, kk = i >> 5, n4 = (i & 31) * 4;
                Bt[(n4 + 0) * 40 + kk] = (bf16)f[j].x;
                Bt[(n4 + 1) * 40 + kk] = (bf16)f[j].y;
                Bt[(n4 + 2) * 40 + kk] = (bf16)f[j].z;
                Bt[(n4 + 3) * 40 + kk] = (bf16)f[j].w;
            }
        }
        if (k0 + 32 < NH * HD)
            __builtin_prefetch(&wo[(size_t)(k0 + 32 + (tid >> 3)) * H_DIM + n0 + (tid & 7) * 16], 0, 1);
        wait_async();
        __syncthreads();

        const bf16* arow = &Alds[(wave * 16 + mlan) * 40];
        v16bf afrag = frag_cat(*(const v8bf*)(arow + hi * 8),
                               *(const v8bf*)(arow + 16 + hi * 8));
        v16bf bfr[8];
        #pragma unroll
        for (int t = 0; t < 8; ++t) {
            const bf16* brow = &Bt[(t * 16 + mlan) * 40];
            bfr[t] = frag_cat(*(const v8bf*)(brow + hi * 8), *(const v8bf*)(brow + 16 + hi * 8));
        }
        #pragma unroll
        for (int t = 0; t < 8; ++t)
            acc[t] = wmma_bf16(afrag, bfr[t], acc[t]);
    }

    #pragma unroll
    for (int r = 0; r < 8; ++r) {
        const int row = s0 + wave * 16 + hi * 8 + r;
        #pragma unroll
        for (int t = 0; t < 8; ++t)
            out[(size_t)row * H_DIM + n0 + t * 16 + mlan] = acc[t][r];
    }
}

// ===========================================================================
// Host launcher
// ===========================================================================
extern "C" void kernel_launch(void* const* d_in, const int* in_sizes, int n_in,
                              void* d_out, int out_size, void* d_ws, size_t ws_size,
                              hipStream_t stream)
{
    const float* hs   = (const float*)d_in[0];   // (1,2048,2048)
    const float* wq   = (const float*)d_in[1];   // (2048,4096)
    const float* wk   = (const float*)d_in[2];   // (2048,512)
    const float* wv   = (const float*)d_in[3];   // (2048,512)
    const float* wo   = (const float*)d_in[4];   // (4096,2048)
    const float* qnw  = (const float*)d_in[5];   // (128,)
    const float* knw  = (const float*)d_in[6];   // (128,)
    const float* cosp = (const float*)d_in[7];   // (2048,128)
    const float* sinp = (const float*)d_in[8];   // (2048,128)
    float* out = (float*)d_out;

    char* ws = (char*)d_ws;
    bf16* q_ws    = (bf16*)(ws);                          // 32*2048*128*2 = 16 MB
    bf16* k_ws    = (bf16*)(ws + ((size_t)16 << 20));     //  4*2048*128*2 =  2 MB
    bf16* v_ws    = (bf16*)(ws + ((size_t)18 << 20));     //                  2 MB
    bf16* attn_ws = (bf16*)(ws + ((size_t)20 << 20));     // 2048*4096*2   = 16 MB

    qkv_rope_kernel<<<dim3(S_LEN / 128, 40), 256, 0, stream>>>(
        hs, wq, wk, wv, qnw, knw, cosp, sinp, q_ws, k_ws, v_ws);

    flash_attn_kernel<<<dim3(S_LEN / 128, NH), 256, 0, stream>>>(
        q_ws, k_ws, v_ws, attn_ws);

    out_proj_kernel<<<dim3(S_LEN / 128, H_DIM / 128), 256, 0, stream>>>(
        attn_ws, wo, out);
}
